// GPT2Attention_79963701117162
// MI455X (gfx1250) — compile-verified, matrix operand order checked by reading, not run on hardware
//
#include <hip/hip_runtime.h>

// ---------------------------------------------------------------------------
// MI455X (gfx1250) GPT-2 attention block, round 2.
//  - one bf16 pre-convert pass (weights/activations), then:
//  - GEMMs + prefill attention on v_wmma_f32_16x16x32_bf16 (fp32 accum)
//  - LDS staging via global_load_async_to_lds_b128 (ASYNCcnt) where no
//    transpose is needed; batched reg loads + b16 scatter for transposes
//  - decode attention: bandwidth-bound VALU sweep of the 1.07GB f32 KV cache
// ---------------------------------------------------------------------------

typedef __bf16 bf16;
typedef __attribute__((ext_vector_type(4)))  bf16  v4bf;
typedef __attribute__((ext_vector_type(8)))  bf16  v8bf;
typedef __attribute__((ext_vector_type(16))) bf16  v16bf;
typedef __attribute__((ext_vector_type(8)))  float v8f;
typedef __attribute__((ext_vector_type(4)))  float v4f;

#define LOG2E 1.44269504088896340736f
#define SCALE 0.08838834764831845f   // 1/sqrt(128)

// --- CDNA5 async global->LDS copy (16B per lane), tracked by ASYNCcnt -------
__device__ __forceinline__ void async_b128(unsigned lds_off, const void* gptr) {
  asm volatile("global_load_async_to_lds_b128 %0, %1, off"
               :: "v"(lds_off), "v"(gptr) : "memory");
}
__device__ __forceinline__ void wait_async0() {
  asm volatile("s_wait_asynccnt 0x0" ::: "memory");
}
// Generic LDS pointer -> 32-bit LDS byte offset (LDS aperture keeps the
// offset in addr[31:0]).
__device__ __forceinline__ unsigned lds_off(const void* p) {
  return (unsigned)(size_t)p;
}

// Load one 16x32 bf16 WMMA operand fragment from an LDS row (K-contiguous).
// 16-bit A layout: lane holds M=lane%16; VGPR0-3 = K[half*8 .. +7],
// VGPR4-7 = K[16+half*8 .. +7] -> two 16B loads at bytes half*16, 32+half*16.
__device__ __forceinline__ v16bf frag_ld(const bf16* rowbase, int half) {
  const v8bf lo = *(const v8bf*)((const char*)rowbase + half * 16);
  const v8bf hi = *(const v8bf*)((const char*)rowbase + 32 + half * 16);
  return __builtin_shufflevector(lo, hi, 0,1,2,3,4,5,6,7,8,9,10,11,12,13,14,15);
}

__device__ __forceinline__ v8f wmma_bf16(v16bf a, v16bf b, v8f c) {
  return __builtin_amdgcn_wmma_f32_16x16x32_bf16(false, a, false, b,
                                                 (short)0, c, false, false);
}

// ---------------------------------------------------------------------------
// Kernel: elementwise f32 -> bf16 (8 elems/thread)
// ---------------------------------------------------------------------------
__global__ __launch_bounds__(256)
void cvt_f32_bf16(const float* __restrict__ in, bf16* __restrict__ out, int n)
{
  const int i = (blockIdx.x * 256 + threadIdx.x) * 8;
  if (i + 8 <= n) {
    v4f a = *(const v4f*)(in + i);
    v4f b = *(const v4f*)(in + i + 4);
    v8bf o = { (bf16)a[0], (bf16)a[1], (bf16)a[2], (bf16)a[3],
               (bf16)b[0], (bf16)b[1], (bf16)b[2], (bf16)b[3] };
    *(v8bf*)(out + i) = o;
  }
}

// ---------------------------------------------------------------------------
// GEMM: C[M,N] = A[M,K](bf16) @ B[K,N](bf16) + bias.  Block tile 128x128x32,
// 8 waves in 2x4 grid, each wave 64x32 (8 fp32 accumulators).
// A tile staged with async copies; B tile transposed via reg-batched stores.
// ---------------------------------------------------------------------------
#define AK 40   // padded LDS K-stride (80B rows, 16B aligned, conflict-free)

template<bool OUT_BF16>
__global__ __launch_bounds__(256)
void gemm_bf16_wmma(const bf16* __restrict__ A, const bf16* __restrict__ Bw,
                    const float* __restrict__ bias,
                    float* __restrict__ Cf, bf16* __restrict__ Cb,
                    int M, int N, int K)
{
  __shared__ bf16 As[2][128 * AK];
  __shared__ bf16 Bs[2][128 * AK];   // stored transposed: [n][k]

  const int tid  = threadIdx.x;
  const int wave = tid >> 5, lane = tid & 31;
  const int wm = wave >> 2, wn = wave & 3;
  const int half = lane >> 4, l16 = lane & 15;
  const int bm = blockIdx.y * 128;
  const int bn = blockIdx.x * 128;

  v8f acc[4][2] = {};

  auto stage = [&](int buf, int k0) {
    // A tile 128x32: async global->LDS, two 16B chunks per thread
#pragma unroll
    for (int i = 0; i < 2; ++i) {
      const int chunk = i * 256 + tid;          // 512 chunks
      const int r = chunk >> 2, c = (chunk & 3) * 8;
      int gr = bm + r; if (gr >= M) gr = M - 1; // clamp (stores guarded)
      async_b128(lds_off(&As[buf][r * AK + c]),
                 A + (size_t)gr * K + k0 + c);
    }
    // B tile 32x128: batch reg loads (MLP), then transpose-scatter to [n][k]
    v8bf t[2];
#pragma unroll
    for (int i = 0; i < 2; ++i) {
      const int chunk = i * 256 + tid;          // 512 chunks
      const int kk = chunk >> 4, n = (chunk & 15) * 8;
      t[i] = *(const v8bf*)(Bw + (size_t)(k0 + kk) * N + bn + n);
    }
#pragma unroll
    for (int i = 0; i < 2; ++i) {
      const int chunk = i * 256 + tid;
      const int kk = chunk >> 4, n = (chunk & 15) * 8;
#pragma unroll
      for (int j = 0; j < 8; ++j)
        Bs[buf][(n + j) * AK + kk] = t[i][j];
    }
  };

  stage(0, 0);
  const int ksteps = K >> 5;
  for (int ks = 0; ks < ksteps; ++ks) {
    wait_async0();          // this buffer's async copies have landed
    __syncthreads();        // ... for every wave in the block
    const int buf = ks & 1;
    if (ks + 1 < ksteps) stage(buf ^ 1, (ks + 1) * 32);

    v16bf af[4], bfr[2];
#pragma unroll
    for (int mf = 0; mf < 4; ++mf)
      af[mf] = frag_ld(&As[buf][(wm * 64 + mf * 16 + l16) * AK], half);
#pragma unroll
    for (int nf = 0; nf < 2; ++nf)
      bfr[nf] = frag_ld(&Bs[buf][(wn * 32 + nf * 16 + l16) * AK], half);

#pragma unroll
    for (int mf = 0; mf < 4; ++mf)
#pragma unroll
      for (int nf = 0; nf < 2; ++nf)
        acc[mf][nf] = wmma_bf16(af[mf], bfr[nf], acc[mf][nf]);
  }

  // Epilogue: C/D layout -> lane holds (M = r + 8*(l>>4), N = l&15) in VGPR r
#pragma unroll
  for (int mf = 0; mf < 4; ++mf)
#pragma unroll
    for (int nf = 0; nf < 2; ++nf) {
      const int col = bn + wn * 32 + nf * 16 + l16;
      const float bv = bias[col];
#pragma unroll
      for (int r = 0; r < 8; ++r) {
        const int row = bm + wm * 64 + mf * 16 + r + 8 * half;
        if (row < M) {
          const float v = acc[mf][nf][r] + bv;
          if (OUT_BF16) Cb[(size_t)row * N + col] = (bf16)v;
          else          Cf[(size_t)row * N + col] = v;
        }
      }
    }
}

// ---------------------------------------------------------------------------
// Prefill flash attention: one block per (b, h, 128-query tile); 8 waves,
// each owns 16 query rows; 64-key tiles; online softmax. All bf16 in.
// ---------------------------------------------------------------------------
#define DPAD 136   // Q/K LDS row stride (272B, 16B aligned, conflict-free)
#define KPAD 72    // P / V^T LDS row stride (144B)

__global__ __launch_bounds__(256)
void prefill_flash_attn(const bf16* __restrict__ qkv, bf16* __restrict__ attn)
{
  __shared__ bf16 Qs[128 * DPAD];
  __shared__ bf16 Ks[64 * DPAD];
  __shared__ bf16 Vts[128 * KPAD];      // V transposed: [d][key]
  __shared__ bf16 Ps[8][16 * KPAD];     // per-wave P tile

  const int qt = blockIdx.x, h = blockIdx.y, b = blockIdx.z;
  const int tid = threadIdx.x, wave = tid >> 5, lane = tid & 31;
  const int half = lane >> 4, l16 = lane & 15;
  const int qrow0 = qt * 128;

  // Stage Q tile: async copies (row-major, no transpose)
#pragma unroll
  for (int i = 0; i < 8; ++i) {
    const int chunk = i * 256 + tid;           // 2048 chunks of 8 elems
    const int r = chunk >> 4, d = (chunk & 15) * 8;
    async_b128(lds_off(&Qs[r * DPAD + d]),
               qkv + (size_t)(b * 1024 + qrow0 + r) * 6144 + h * 128 + d);
  }

  float m_i[8], l_i[8];
  v8f o[8] = {};
#pragma unroll
  for (int r = 0; r < 8; ++r) { m_i[r] = -1e30f; l_i[r] = 0.f; }

  const int nkt = qt * 2 + 2;   // causal: only key tiles at/below the diagonal
  for (int kt = 0; kt < nkt; ++kt) {
    __syncthreads();            // all waves done reading previous K/V tiles
    // K tile [64 keys][128 d]: async copies
#pragma unroll
    for (int i = 0; i < 4; ++i) {
      const int chunk = i * 256 + tid;         // 1024 chunks
      const int r = chunk >> 4, d = (chunk & 15) * 8;
      async_b128(lds_off(&Ks[r * DPAD + d]),
                 qkv + (size_t)(b * 1024 + kt * 64 + r) * 6144 + 2048 + h * 128 + d);
    }
    // V tile transposed [128 d][64 keys]: batch reg loads then b16 scatter
    v8bf tv[4];
#pragma unroll
    for (int i = 0; i < 4; ++i) {
      const int chunk = i * 256 + tid;         // 1024 chunks
      const int r = chunk >> 4, d = (chunk & 15) * 8;
      tv[i] = *(const v8bf*)(qkv + (size_t)(b * 1024 + kt * 64 + r) * 6144 + 4096 + h * 128 + d);
    }
#pragma unroll
    for (int i = 0; i < 4; ++i) {
      const int chunk = i * 256 + tid;
      const int r = chunk >> 4, d = (chunk & 15) * 8;
#pragma unroll
      for (int j = 0; j < 8; ++j) Vts[(d + j) * KPAD + r] = tv[i][j];
    }
    wait_async0();
    __syncthreads();

    // S = Q K^T : 16x64 per wave, K-dim = D = 128 (4 WMMA k-steps)
    v8f s[4] = {};
#pragma unroll
    for (int dk = 0; dk < 4; ++dk) {
      v16bf aq = frag_ld(&Qs[(wave * 16 + l16) * DPAD + dk * 32], half);
#pragma unroll
      for (int nt = 0; nt < 4; ++nt) {
        v16bf bk = frag_ld(&Ks[(nt * 16 + l16) * DPAD + dk * 32], half);
        s[nt] = wmma_bf16(aq, bk, s[nt]);
      }
    }

    // Scale, causal mask, online softmax (row stats in VGPR r, row r+8*half)
    float rmax[8];
#pragma unroll
    for (int r = 0; r < 8; ++r) {
      const int qrow = qrow0 + wave * 16 + r + 8 * half;
      float mx = -1e30f;
#pragma unroll
      for (int nt = 0; nt < 4; ++nt) {
        const int key = kt * 64 + nt * 16 + l16;
        float v = s[nt][r] * SCALE;
        v = (key <= qrow) ? v : -1e30f;
        s[nt][r] = v;
        mx = fmaxf(mx, v);
      }
#pragma unroll
      for (int off = 1; off < 16; off <<= 1)
        mx = fmaxf(mx, __shfl_xor(mx, off, 32));
      rmax[r] = mx;
    }
#pragma unroll
    for (int r = 0; r < 8; ++r) {
      const float newm  = fmaxf(m_i[r], rmax[r]);
      const float alpha = exp2f((m_i[r] - newm) * LOG2E);
      float rsum = 0.f;
#pragma unroll
      for (int nt = 0; nt < 4; ++nt) {
        const float p = exp2f((s[nt][r] - newm) * LOG2E);
        s[nt][r] = p;
        rsum += p;
      }
#pragma unroll
      for (int off = 1; off < 16; off <<= 1)
        rsum += __shfl_xor(rsum, off, 32);
      l_i[r] = l_i[r] * alpha + rsum;
      m_i[r] = newm;
#pragma unroll
      for (int nt = 0; nt < 8; ++nt) o[nt][r] *= alpha;
    }

    // P (C/D layout) -> per-wave LDS, re-read as A fragments (same-wave DS
    // ops are in-order; no cross-wave barrier needed)
#pragma unroll
    for (int r = 0; r < 8; ++r)
#pragma unroll
      for (int nt = 0; nt < 4; ++nt)
        Ps[wave][(r + 8 * half) * KPAD + nt * 16 + l16] = (bf16)s[nt][r];

    // O += P @ V : K-dim = 64 keys (2 WMMA k-steps), N = 128
#pragma unroll
    for (int kk = 0; kk < 2; ++kk) {
      v16bf ap = frag_ld(&Ps[wave][l16 * KPAD + kk * 32], half);
#pragma unroll
      for (int nt = 0; nt < 8; ++nt) {
        v16bf bv = frag_ld(&Vts[(nt * 16 + l16) * KPAD + kk * 32], half);
        o[nt] = wmma_bf16(ap, bv, o[nt]);
      }
    }
  }

  // Epilogue: O / l_i -> attn_bf[(b*P+row), h*128 + d]
#pragma unroll
  for (int r = 0; r < 8; ++r) {
    const float inv = 1.0f / l_i[r];
    const size_t orow = (size_t)(b * 1024 + qrow0 + wave * 16 + r + 8 * half);
#pragma unroll
    for (int nt = 0; nt < 8; ++nt)
      attn[orow * 2048 + h * 128 + nt * 16 + l16] = (bf16)(o[nt][r] * inv);
  }
}

// ---------------------------------------------------------------------------
// Decode attention: pure bandwidth (1.07GB f32 KV sweep). One block per
// (g,h); each wave owns key pairs (2x unrolled online softmax, 4 b128 loads
// in flight per iteration); lane holds 4 of 128 dims.
// ---------------------------------------------------------------------------
__device__ __forceinline__ v4f bf4_to_f4(v4bf h) {
  v4f r = { (float)h[0], (float)h[1], (float)h[2], (float)h[3] };
  return r;
}

__global__ __launch_bounds__(256)
void decode_attn(const bf16* __restrict__ qkv, const float* __restrict__ pk,
                 const float* __restrict__ pv, bf16* __restrict__ attn)
{
  const int gh = blockIdx.x;
  const int g = gh >> 4, h = gh & 15;
  const int tid = threadIdx.x, wave = tid >> 5, lane = tid & 31;

  __shared__ float sm[8], sl[8];
  __shared__ float sacc[8][128];

  const size_t qrow = (size_t)(2048 + g) * 6144;
  const v4f q = bf4_to_f4(*(const v4bf*)(qkv + qrow + h * 128 + lane * 4));
  const size_t kvbase = (size_t)(g * 16 + h) * 2048 * 128;

  float m = -1e30f, l = 0.f;
  v4f acc = {0.f, 0.f, 0.f, 0.f};

  for (int it = 0; it < 128; ++it) {
    const int k0 = it * 16 + wave * 2;                 // keys k0, k0+1
    const float* kp = pk + kvbase + (size_t)k0 * 128;
    const float* vp = pv + kvbase + (size_t)k0 * 128;
    const v4f ka = *(const v4f*)(kp + lane * 4);
    const v4f kb = *(const v4f*)(kp + 128 + lane * 4);
    const v4f va = *(const v4f*)(vp + lane * 4);
    const v4f vb = *(const v4f*)(vp + 128 + lane * 4);
    float d0 = q[0]*ka[0] + q[1]*ka[1] + q[2]*ka[2] + q[3]*ka[3];
    float d1 = q[0]*kb[0] + q[1]*kb[1] + q[2]*kb[2] + q[3]*kb[3];
#pragma unroll
    for (int off = 1; off < 32; off <<= 1) {
      d0 += __shfl_xor(d0, off, 32);
      d1 += __shfl_xor(d1, off, 32);
    }
    const float s0 = d0 * SCALE, s1 = d1 * SCALE;
    const float newm  = fmaxf(m, fmaxf(s0, s1));
    const float alpha = exp2f((m - newm) * LOG2E);
    const float p0 = exp2f((s0 - newm) * LOG2E);
    const float p1 = exp2f((s1 - newm) * LOG2E);
    l = l * alpha + p0 + p1;
    m = newm;
    acc = acc * alpha + va * p0 + vb * p1;
  }
  // Tail: freshly appended key/value at position 2048 (bf16 from qkv)
  if (wave == 0) {
    const v4f kc = bf4_to_f4(*(const v4bf*)(qkv + qrow + 2048 + h * 128 + lane * 4));
    const v4f vc = bf4_to_f4(*(const v4bf*)(qkv + qrow + 4096 + h * 128 + lane * 4));
    float d = q[0]*kc[0] + q[1]*kc[1] + q[2]*kc[2] + q[3]*kc[3];
#pragma unroll
    for (int off = 1; off < 32; off <<= 1) d += __shfl_xor(d, off, 32);
    const float s = d * SCALE;
    const float newm  = fmaxf(m, s);
    const float alpha = exp2f((m - newm) * LOG2E);
    const float p     = exp2f((s - newm) * LOG2E);
    l = l * alpha + p;
    m = newm;
    acc = acc * alpha + vc * p;
  }

  if (lane == 0) { sm[wave] = m; sl[wave] = l; }
  *(v4f*)(&sacc[wave][lane * 4]) = acc;
  __syncthreads();

  if (tid < 128) {
    float M = -1e30f;
#pragma unroll
    for (int w = 0; w < 8; ++w) M = fmaxf(M, sm[w]);
    float L = 0.f, a = 0.f;
#pragma unroll
    for (int w = 0; w < 8; ++w) {
      const float sc = exp2f((sm[w] - M) * LOG2E);
      L += sl[w] * sc;
      a += sacc[w][tid] * sc;
    }
    attn[(size_t)(2048 + g) * 2048 + h * 128 + tid] = (bf16)(a / L);
  }
}

// ---------------------------------------------------------------------------
// Launch: cvt x3 -> qkv GEMM(bf16 out) -> {prefill, decode} -> proj GEMM(f32)
// ---------------------------------------------------------------------------
extern "C" void kernel_launch(void* const* d_in, const int* in_sizes, int n_in,
                              void* d_out, int out_size, void* d_ws, size_t ws_size,
                              hipStream_t stream)
{
  const float* hs = (const float*)d_in[0];
  const float* pk = (const float*)d_in[1];
  const float* pv = (const float*)d_in[2];
  const float* w1 = (const float*)d_in[3];
  const float* b1 = (const float*)d_in[4];
  const float* w2 = (const float*)d_in[5];
  const float* b2 = (const float*)d_in[6];
  float* out = (float*)d_out;

  bf16* hs_bf  = (bf16*)d_ws;                        //  8.5 MB
  bf16* w1_bf  = hs_bf  + (size_t)2080 * 2048;       // 25.2 MB
  bf16* w2_bf  = w1_bf  + (size_t)2048 * 6144;       //  8.4 MB
  bf16* qkvbf  = w2_bf  + (size_t)2048 * 2048;       // 25.6 MB
  bf16* attnbf = qkvbf  + (size_t)2080 * 6144;       //  8.5 MB  (total ~76 MB)

  const dim3 blk(256);
  cvt_f32_bf16<<<dim3(2080), blk, 0, stream>>>(hs, hs_bf, 2080 * 2048);
  cvt_f32_bf16<<<dim3(6144), blk, 0, stream>>>(w1, w1_bf, 2048 * 6144);
  cvt_f32_bf16<<<dim3(2048), blk, 0, stream>>>(w2, w2_bf, 2048 * 2048);

  gemm_bf16_wmma<true ><<<dim3(48, 17), blk, 0, stream>>>(
      hs_bf, w1_bf, b1, nullptr, qkvbf, 2080, 6144, 2048);
  prefill_flash_attn<<<dim3(8, 16, 2), blk, 0, stream>>>(qkvbf, attnbf);
  decode_attn<<<dim3(512), blk, 0, stream>>>(qkvbf, pk, pv, attnbf);
  gemm_bf16_wmma<false><<<dim3(16, 17), blk, 0, stream>>>(
      attnbf, w2_bf, b2, out, nullptr, 2080, 2048, 2048);
}
